// MultiHeadAttention_31731218383195
// MI455X (gfx1250) — matrix operand order repaired and checked by reading, not compiled, not run on hardware
//
#include <hip/hip_runtime.h>
#include <hip/hip_bf16.h>

typedef __attribute__((ext_vector_type(16))) __bf16 bfx16;
typedef __attribute__((ext_vector_type(8)))  __bf16 bfx8;
typedef __attribute__((ext_vector_type(4)))  __bf16 bfx4;
typedef __attribute__((ext_vector_type(8)))  float  fx8;

union ABfrag { bfx16 v; bfx8 h[2]; };

#define D_MODEL 768
#define SEQ     2048
#define NHEAD   12
#define HDIM    64

// ---------------------------------------------------------------------------
// fp32 -> bf16 conversion (vectorized, n multiple of 4)
// ---------------------------------------------------------------------------
__global__ __launch_bounds__(256) void cvt_f32_bf16(const float4* __restrict__ in,
                                                    bfx4* __restrict__ out, int n4) {
    int i = blockIdx.x * 256 + threadIdx.x;
    if (i < n4) {
        float4 f = in[i];
        bfx4 o = { (__bf16)f.x, (__bf16)f.y, (__bf16)f.z, (__bf16)f.w };
        out[i] = o;
    }
}

// ---------------------------------------------------------------------------
// C[m,n] = sum_k X[m,k] * W[n,k]   (torch Linear: X @ W^T), bf16 in, bf16 out
// Workgroup tile 128x64, 8 waves, each wave 32x32 (2x2 WMMA tiles).
// ---------------------------------------------------------------------------
__global__ __launch_bounds__(256) void gemm_proj_bf16(
    const __bf16* __restrict__ X, const __bf16* __restrict__ W,
    __bf16* __restrict__ C, long crs, long ccs)
{
    const int lane = threadIdx.x & 31;
    const int w    = threadIdx.x >> 5;
    const int lh   = lane >> 4;
    const int ll   = lane & 15;
    const int m0 = blockIdx.x * 128 + (w & 3) * 32;
    const int n0 = blockIdx.y * 64  + (w >> 2) * 32;

    fx8 acc[2][2] = {};
    for (int k0 = 0; k0 < D_MODEL; k0 += 32) {
        if (k0 + 32 < D_MODEL) {   // stream-ahead hint -> global_prefetch_b8
            __builtin_prefetch(X + (long)(m0 + lane) * D_MODEL + k0 + 32);
            __builtin_prefetch(W + (long)(n0 + lane) * D_MODEL + k0 + 32);
        }
        ABfrag a[2]; bfx16 bm[2];
#pragma unroll
        for (int t = 0; t < 2; ++t) {
            const __bf16* ap = X + (long)(m0 + t * 16 + ll) * D_MODEL + k0 + lh * 8;
            a[t].h[0] = *(const bfx8*)(ap);
            a[t].h[1] = *(const bfx8*)(ap + 16);
            const __bf16* bp = W + (long)(n0 + t * 16 + ll) * D_MODEL + k0 + lh * 8;
            bm[t] = __builtin_shufflevector(*(const bfx8*)(bp), *(const bfx8*)(bp + 16), 0, 1, 2, 3, 4, 5, 6, 7, 8, 9, 10, 11, 12, 13, 14, 15);
        }
#pragma unroll
        for (int i = 0; i < 2; ++i)
#pragma unroll
            for (int j = 0; j < 2; ++j)
                acc[i][j] = __builtin_amdgcn_wmma_f32_16x16x32_bf16(
                    false, a[i].v, false, bm[j], (short)0, acc[i][j], false, false);
    }
#pragma unroll
    for (int i = 0; i < 2; ++i)
#pragma unroll
        for (int j = 0; j < 2; ++j)
#pragma unroll
            for (int e = 0; e < 8; ++e) {
                int m = m0 + i * 16 + e + lh * 8;
                int n = n0 + j * 16 + ll;
                C[(long)m * crs + (long)n * ccs] = (__bf16)acc[i][j][e];
            }
}

// ---------------------------------------------------------------------------
// Output projection + bias + residual -> fp32 (pre-LayerNorm), M = 4096
// ---------------------------------------------------------------------------
__global__ __launch_bounds__(256) void gemm_out_resid(
    const __bf16* __restrict__ X, const __bf16* __restrict__ W,
    const float* __restrict__ bias, const float* __restrict__ resid,
    float* __restrict__ C)
{
    const int lane = threadIdx.x & 31;
    const int w    = threadIdx.x >> 5;
    const int lh   = lane >> 4;
    const int ll   = lane & 15;
    const int m0 = blockIdx.x * 128 + (w & 3) * 32;
    const int n0 = blockIdx.y * 64  + (w >> 2) * 32;

    fx8 acc[2][2] = {};
    for (int k0 = 0; k0 < D_MODEL; k0 += 32) {
        if (k0 + 32 < D_MODEL) {
            __builtin_prefetch(X + (long)(m0 + lane) * D_MODEL + k0 + 32);
            __builtin_prefetch(W + (long)(n0 + lane) * D_MODEL + k0 + 32);
        }
        ABfrag a[2]; bfx16 bm[2];
#pragma unroll
        for (int t = 0; t < 2; ++t) {
            const __bf16* ap = X + (long)(m0 + t * 16 + ll) * D_MODEL + k0 + lh * 8;
            a[t].h[0] = *(const bfx8*)(ap);
            a[t].h[1] = *(const bfx8*)(ap + 16);
            const __bf16* bp = W + (long)(n0 + t * 16 + ll) * D_MODEL + k0 + lh * 8;
            bm[t] = __builtin_shufflevector(*(const bfx8*)(bp), *(const bfx8*)(bp + 16), 0, 1, 2, 3, 4, 5, 6, 7, 8, 9, 10, 11, 12, 13, 14, 15);
        }
#pragma unroll
        for (int i = 0; i < 2; ++i)
#pragma unroll
            for (int j = 0; j < 2; ++j)
                acc[i][j] = __builtin_amdgcn_wmma_f32_16x16x32_bf16(
                    false, a[i].v, false, bm[j], (short)0, acc[i][j], false, false);
    }
#pragma unroll
    for (int i = 0; i < 2; ++i)
#pragma unroll
        for (int j = 0; j < 2; ++j)
#pragma unroll
            for (int e = 0; e < 8; ++e) {
                int m = m0 + i * 16 + e + lh * 8;
                int n = n0 + j * 16 + ll;
                long idx = (long)m * D_MODEL + n;
                C[idx] = acc[i][j][e] + bias[n] + resid[idx];
            }
}

// ---------------------------------------------------------------------------
// Fused flash attention. grid = (L/64, H, B), 128 threads = 4 waves.
// K/V chunks (64x64 bf16 each) are staged into LDS by the whole workgroup with
// async DMA (global_load_async_to_lds_b128, ASYNCcnt), double-buffered so the
// DMA of chunk i+1 overlaps the WMMA work on chunk i.
// ---------------------------------------------------------------------------
#define KSTR 72
__global__ __launch_bounds__(128) void attn_flash(
    const __bf16* __restrict__ Q, const __bf16* __restrict__ K,
    const __bf16* __restrict__ Vt, const int* __restrict__ amask,
    __bf16* __restrict__ O)
{
    __shared__ __align__(16) __bf16 Ks[2][64][KSTR];     // K chunk, rows = s
    __shared__ __align__(16) __bf16 Vs[2][64][KSTR];     // V chunk, rows = d (transposed V)
    __shared__ __align__(16) __bf16 Plds[4][16][KSTR];   // per-wave P staging
    const int tid  = threadIdx.x;
    const int lane = tid & 31;
    const int w    = tid >> 5;
    const int lh   = lane >> 4;
    const int ll   = lane & 15;
    const int b  = blockIdx.z;
    const int h  = blockIdx.y;
    const int l0 = blockIdx.x * 64 + w * 16;
    const long qkbase = (long)b * SEQ * D_MODEL + h * HDIM;
    const long vtbase = ((long)b * D_MODEL + h * HDIM) * SEQ;

    // async cooperative stage of one 64-col chunk of K and V into LDS buffer `buf`
    // 64 rows x 128B per matrix = 512 x 16B units; 128 threads x 4 units each.
    auto issue_chunk = [&](int s0c, int buf) {
#pragma unroll
        for (int j = 0; j < 4; ++j) {
            int u   = tid + 128 * j;
            int row = u >> 3;             // 8 x 16B units per 64-elt row
            int c8  = (u & 7) * 8;
            unsigned kl = (unsigned)(uintptr_t)&Ks[buf][row][c8];
            unsigned long long kg =
                (unsigned long long)(uintptr_t)(K + qkbase + (long)(s0c + row) * D_MODEL + c8);
            asm volatile("global_load_async_to_lds_b128 %0, %1, off"
                         :: "v"(kl), "v"(kg) : "memory");
            unsigned vl = (unsigned)(uintptr_t)&Vs[buf][row][c8];
            unsigned long long vg =
                (unsigned long long)(uintptr_t)(Vt + vtbase + (long)row * SEQ + s0c + c8);
            asm volatile("global_load_async_to_lds_b128 %0, %1, off"
                         :: "v"(vl), "v"(vg) : "memory");
        }
    };

    // Q fragments for this wave's 16 rows (reused across all S chunks)
    ABfrag qa[2];
#pragma unroll
    for (int ks = 0; ks < 2; ++ks) {
        const __bf16* qp = Q + qkbase + (long)(l0 + ll) * D_MODEL + ks * 32 + lh * 8;
        qa[ks].h[0] = *(const bfx8*)qp;
        qa[ks].h[1] = *(const bfx8*)(qp + 16);
    }

    float mrun[8], lrun[8];
    fx8 accv[4] = {};
#pragma unroll
    for (int r = 0; r < 8; ++r) { mrun[r] = -3.0e38f; lrun[r] = 0.0f; }
    const float scale = 0.125f;  // 1/sqrt(64)

    issue_chunk(0, 0);                       // preload chunk 0
    const int NCH = SEQ / 64;
    for (int ic = 0; ic < NCH; ++ic) {
        const int s0  = ic * 64;
        const int cur = ic & 1;
        if (ic + 1 < NCH) {
            issue_chunk(s0 + 64, cur ^ 1);   // DMA next chunk into other buffer
            // wait until only the 8 newest async ops (next chunk) remain
            asm volatile("s_wait_asynccnt 0x8" ::: "memory");
        } else {
            asm volatile("s_wait_asynccnt 0x0" ::: "memory");
        }
        __syncthreads();                     // all waves' chunk-i DMA visible

        // ---- scores: 16x64 chunk via 4 n-tiles x 2 k-steps of WMMA (B from LDS)
        fx8 st[4];
#pragma unroll
        for (int nt = 0; nt < 4; ++nt) {
            fx8 s = {};
#pragma unroll
            for (int ks = 0; ks < 2; ++ks) {
                ABfrag bm;
                const __bf16* kp = &Ks[cur][nt * 16 + ll][ks * 32 + lh * 8];
                bm.h[0] = *(const bfx8*)kp;
                bm.h[1] = *(const bfx8*)(kp + 16);
                s = __builtin_amdgcn_wmma_f32_16x16x32_bf16(
                        false, qa[ks].v, false, bm.v, (short)0, s, false, false);
            }
            int scol = s0 + nt * 16 + ll;
            float madd = (1.0f - (float)amask[b * SEQ + scol]) * -1000000.0f;
#pragma unroll
            for (int e = 0; e < 8; ++e) s[e] = s[e] * scale + madd;
            st[nt] = s;
        }
        // ---- online softmax (rows live in 16-lane halves of the C layout)
#pragma unroll
        for (int r = 0; r < 8; ++r) {
            float cm = fmaxf(fmaxf(st[0][r], st[1][r]), fmaxf(st[2][r], st[3][r]));
            cm = fmaxf(cm, __shfl_xor(cm, 1, 16));
            cm = fmaxf(cm, __shfl_xor(cm, 2, 16));
            cm = fmaxf(cm, __shfl_xor(cm, 4, 16));
            cm = fmaxf(cm, __shfl_xor(cm, 8, 16));
            float mnew = fmaxf(mrun[r], cm);
            float corr = __expf(mrun[r] - mnew);
            mrun[r] = mnew;
            lrun[r] *= corr;
#pragma unroll
            for (int dt = 0; dt < 4; ++dt) accv[dt][r] *= corr;
            float rs = 0.0f;
#pragma unroll
            for (int nt = 0; nt < 4; ++nt) {
                float p = __expf(st[nt][r] - mnew);
                st[nt][r] = p;
                rs += p;
            }
            rs += __shfl_xor(rs, 1, 16);
            rs += __shfl_xor(rs, 2, 16);
            rs += __shfl_xor(rs, 4, 16);
            rs += __shfl_xor(rs, 8, 16);
            lrun[r] += rs;
        }
        // ---- stage P (C layout -> row-major bf16) through this wave's LDS slice
#pragma unroll
        for (int nt = 0; nt < 4; ++nt)
#pragma unroll
            for (int e = 0; e < 8; ++e)
                Plds[w][e + lh * 8][nt * 16 + ll] = (__bf16)st[nt][e];
        // ---- accumulate P @ V (A from Plds, B from staged V chunk)
#pragma unroll
        for (int ks = 0; ks < 2; ++ks) {
            ABfrag pa;
            const __bf16* pp = &Plds[w][ll][ks * 32 + lh * 8];
            pa.h[0] = *(const bfx8*)pp;
            pa.h[1] = *(const bfx8*)(pp + 16);
#pragma unroll
            for (int dt = 0; dt < 4; ++dt) {
                ABfrag bm;
                const __bf16* vp = &Vs[cur][dt * 16 + ll][ks * 32 + lh * 8];
                bm.h[0] = *(const bfx8*)vp;
                bm.h[1] = *(const bfx8*)(vp + 16);
                accv[dt] = __builtin_amdgcn_wmma_f32_16x16x32_bf16(
                               false, pa.v, false, bm.v, (short)0, accv[dt], false, false);
            }
        }
        __syncthreads();                     // buffer `cur` free for DMA of chunk i+2
    }
    // ---- finalize: O = acc / l
#pragma unroll
    for (int e = 0; e < 8; ++e) {
        float inv = 1.0f / lrun[e];
        int row = l0 + e + lh * 8;
#pragma unroll
        for (int dt = 0; dt < 4; ++dt)
            O[((long)b * SEQ + row) * D_MODEL + h * HDIM + dt * 16 + ll] =
                (__bf16)(accv[dt][e] * inv);
    }
}

// ---------------------------------------------------------------------------
// LayerNorm over 768 features, one 256-thread block per row.
// ---------------------------------------------------------------------------
__global__ __launch_bounds__(256) void layernorm_rows(
    const float* __restrict__ X, const float* __restrict__ gamma,
    const float* __restrict__ beta, float* __restrict__ out)
{
    __shared__ float red[8];
    const int row = blockIdx.x;
    const int tid = threadIdx.x;
    const float* x = X + (long)row * D_MODEL;
    float* o = out + (long)row * D_MODEL;

    float v0 = x[tid], v1 = x[tid + 256], v2 = x[tid + 512];
    float s = v0 + v1 + v2;
#pragma unroll
    for (int m = 1; m < 32; m <<= 1) s += __shfl_xor(s, m, 32);
    if ((tid & 31) == 0) red[tid >> 5] = s;
    __syncthreads();
    float mean = 0.0f;
#pragma unroll
    for (int i = 0; i < 8; ++i) mean += red[i];
    mean *= (1.0f / 768.0f);
    __syncthreads();

    float d0 = v0 - mean, d1 = v1 - mean, d2 = v2 - mean;
    float vs = d0 * d0 + d1 * d1 + d2 * d2;
#pragma unroll
    for (int m = 1; m < 32; m <<= 1) vs += __shfl_xor(vs, m, 32);
    if ((tid & 31) == 0) red[tid >> 5] = vs;
    __syncthreads();
    float var = 0.0f;
#pragma unroll
    for (int i = 0; i < 8; ++i) var += red[i];
    var *= (1.0f / 768.0f);
    float rstd = rsqrtf(var + 1e-5f);

    o[tid]       = d0 * rstd * gamma[tid]       + beta[tid];
    o[tid + 256] = d1 * rstd * gamma[tid + 256] + beta[tid + 256];
    o[tid + 512] = d2 * rstd * gamma[tid + 512] + beta[tid + 512];
}

// ---------------------------------------------------------------------------
extern "C" void kernel_launch(void* const* d_in, const int* in_sizes, int n_in,
                              void* d_out, int out_size, void* d_ws, size_t ws_size,
                              hipStream_t stream) {
    const float* q     = (const float*)d_in[0];
    const float* k     = (const float*)d_in[1];
    const float* v     = (const float*)d_in[2];
    const int*   amask = (const int*)  d_in[3];
    const float* Wq    = (const float*)d_in[4];
    const float* Wk    = (const float*)d_in[5];
    const float* Wv    = (const float*)d_in[6];
    const float* Wo    = (const float*)d_in[7];
    const float* bias  = (const float*)d_in[8];
    const float* gamma = (const float*)d_in[9];
    const float* beta  = (const float*)d_in[10];

    const long NACT = 2L * SEQ * D_MODEL;        // 3,145,728
    const long NW   = (long)D_MODEL * D_MODEL;   //   589,824
    const size_t ACT_B = NACT * 2;               // 6,291,456 bytes
    const size_t W_B   = NW * 2;                 // 1,179,648 bytes

    char* ws = (char*)d_ws;
    __bf16* qb  = (__bf16*)(ws);
    __bf16* kb  = (__bf16*)(ws + ACT_B);
    __bf16* vb  = (__bf16*)(ws + 2 * ACT_B);
    __bf16* wqb = (__bf16*)(ws + 3 * ACT_B);
    __bf16* wkb = (__bf16*)(ws + 3 * ACT_B + W_B);
    __bf16* wvb = (__bf16*)(ws + 3 * ACT_B + 2 * W_B);
    __bf16* wob = (__bf16*)(ws + 3 * ACT_B + 3 * W_B);
    __bf16* Qp  = (__bf16*)(ws + 3 * ACT_B + 4 * W_B);
    __bf16* Kp  = (__bf16*)(ws + 4 * ACT_B + 4 * W_B);
    __bf16* Vt  = (__bf16*)(ws + 5 * ACT_B + 4 * W_B);
    __bf16* Ob  = (__bf16*)(ws);                 // reuse qb region
    float*  Xf  = (float*) (ws + ACT_B);         // reuse kb+vb regions (fp32)

    // 1) bf16 conversions
    cvt_f32_bf16<<<(NACT / 4 + 255) / 256, 256, 0, stream>>>((const float4*)q, (bfx4*)qb, NACT / 4);
    cvt_f32_bf16<<<(NACT / 4 + 255) / 256, 256, 0, stream>>>((const float4*)k, (bfx4*)kb, NACT / 4);
    cvt_f32_bf16<<<(NACT / 4 + 255) / 256, 256, 0, stream>>>((const float4*)v, (bfx4*)vb, NACT / 4);
    cvt_f32_bf16<<<(NW / 4 + 255) / 256, 256, 0, stream>>>((const float4*)Wq, (bfx4*)wqb, NW / 4);
    cvt_f32_bf16<<<(NW / 4 + 255) / 256, 256, 0, stream>>>((const float4*)Wk, (bfx4*)wkb, NW / 4);
    cvt_f32_bf16<<<(NW / 4 + 255) / 256, 256, 0, stream>>>((const float4*)Wv, (bfx4*)wvb, NW / 4);
    cvt_f32_bf16<<<(NW / 4 + 255) / 256, 256, 0, stream>>>((const float4*)Wo, (bfx4*)wob, NW / 4);

    // 2) Q/K/V projections (per batch; V stored transposed (B,768,S))
    dim3 pgrid(SEQ / 128, D_MODEL / 64);
    for (int bb = 0; bb < 2; ++bb) {
        long ao = (long)bb * SEQ * D_MODEL;
        gemm_proj_bf16<<<pgrid, 256, 0, stream>>>(qb + ao, wqb, Qp + ao, D_MODEL, 1);
        gemm_proj_bf16<<<pgrid, 256, 0, stream>>>(kb + ao, wkb, Kp + ao, D_MODEL, 1);
        gemm_proj_bf16<<<pgrid, 256, 0, stream>>>(vb + ao, wvb,
                                                  Vt + (long)bb * D_MODEL * SEQ, 1, SEQ);
    }

    // 3) fused flash attention (async double-buffered K/V staging)
    attn_flash<<<dim3(SEQ / 64, NHEAD, 2), 128, 0, stream>>>(Qp, Kp, Vt, amask, Ob);

    // 4) output projection + bias + residual (fp32)
    gemm_out_resid<<<dim3(2 * SEQ / 128, D_MODEL / 64), 256, 0, stream>>>(
        Ob, wob, bias, q, Xf);

    // 5) LayerNorm -> d_out
    layernorm_rows<<<2 * SEQ, 256, 0, stream>>>(Xf, gamma, beta, (float*)d_out);
}
